// Block_34703335752396
// MI455X (gfx1250) — matrix operand order repaired and checked by reading, not compiled, not run on hardware
//
#include <hip/hip_runtime.h>
#include <hip/hip_bf16.h>
#include <math.h>

typedef _Float16 f16;
typedef __attribute__((ext_vector_type(16))) _Float16 v16h;
typedef __attribute__((ext_vector_type(8)))  float    v8f;

#define NROWS 16384   // B*T
#define DMODEL 512
#define NHEAD 16
#define HDIM 32
#define NEXP 4
#define FDIM 2048
#define KMOE 8192     // E*F
#define CHUNK 2048
#define NCHUNK 8

union FragAB { v16h v; unsigned u[8]; f16 h[16]; };

// ---------------- weight repack / convert ----------------
__global__ __launch_bounds__(256) void cvt_f32_f16(const float* __restrict__ s,
                                                   f16* __restrict__ d, int n) {
  int i = blockIdx.x * 256 + threadIdx.x;
  if (i < n) d[i] = (f16)s[i];
}

// Wqkv[d][c], c<512:q, <1024:k, else v ; w[h][d][e] with h=cc>>5, e=cc&31
__global__ __launch_bounds__(256) void cvt_qkv(const float* __restrict__ wq,
                                               const float* __restrict__ wk,
                                               const float* __restrict__ wv,
                                               f16* __restrict__ dst) {
  int i = blockIdx.x * 256 + threadIdx.x;           // over 512*1536
  int d = i / 1536, c = i % 1536;
  const float* w = (c < 512) ? wq : (c < 1024 ? wk : wv);
  int cc = c & 511;
  int h = cc >> 5, e = cc & 31;
  dst[i] = (f16)w[((size_t)h * DMODEL + d) * HDIM + e];
}

// W1cat[d][e*F+f] = w1[e][d][f]
__global__ __launch_bounds__(256) void cvt_w1(const float* __restrict__ w1,
                                              f16* __restrict__ dst) {
  int i = blockIdx.x * 256 + threadIdx.x;           // over 512*8192
  int d = i >> 13, c = i & 8191;
  int e = c >> 11, f = c & 2047;
  dst[i] = (f16)w1[((size_t)e * DMODEL + d) * FDIM + f];
}

// ---------------- layernorm (block = row) ----------------
__global__ __launch_bounds__(256) void ln_kernel(const float* __restrict__ x,
                                                 const float* __restrict__ g,
                                                 const float* __restrict__ b,
                                                 f16* __restrict__ out) {
  __shared__ float red[256];
  int tid = threadIdx.x;
  const float* xr = x + (size_t)blockIdx.x * DMODEL;
  float v0 = xr[tid], v1 = xr[tid + 256];
  red[tid] = v0 + v1; __syncthreads();
  for (int s = 128; s > 0; s >>= 1) { if (tid < s) red[tid] += red[tid + s]; __syncthreads(); }
  float mean = red[0] * (1.0f / 512.0f); __syncthreads();
  float d0 = v0 - mean, d1 = v1 - mean;
  red[tid] = d0 * d0 + d1 * d1; __syncthreads();
  for (int s = 128; s > 0; s >>= 1) { if (tid < s) red[tid] += red[tid + s]; __syncthreads(); }
  float rs = rsqrtf(red[0] * (1.0f / 512.0f) + 1e-5f);
  f16* orow = out + (size_t)blockIdx.x * DMODEL;
  orow[tid]       = (f16)(d0 * rs * g[tid]       + b[tid]);
  orow[tid + 256] = (f16)(d1 * rs * g[tid + 256] + b[tid + 256]);
}

// ---------------- tiled WMMA GEMM, double-buffered, mode-switched epilogue --
// C[M,N] = A[M,K](f16) * B[K,N](f16); block tile 128x64, K-slab 32, 8 waves,
// each wave a 32x32 quadrant (2x2 WMMA 16x16x32 tiles). Software-pipelined:
// global->regs for slab s+1 issued before computing slab s; regs->LDS into the
// alternate buffer after the WMMAs; one barrier per slab.
// mode 0: outh = f16(acc)
// mode 1: outf = acc + bias[col] + resid[row,col]
// mode 2: outh = f16( gelu(acc + bias[col]) * combine[row*4 + (col>>11)] )
// mode 3: outf = acc + sum_e combine[row*4+e]*b2[e*512+col] + resid[row,col]
__global__ __launch_bounds__(256)
void gemm_wmma(const f16* __restrict__ A, int lda,
               const f16* __restrict__ B, int ldb,
               int Kdim, int mode,
               f16* __restrict__ outh, float* __restrict__ outf, int ldo,
               const float* __restrict__ bias,
               const float* __restrict__ resid,
               const float* __restrict__ combine,
               const float* __restrict__ b2) {
  __shared__ __align__(16) f16 As[2][128 * 40];
  __shared__ __align__(16) f16 Bs[2][64 * 40];
  const int tid  = threadIdx.x;
  const int lane = tid & 31;
  const int wid  = tid >> 5;
  const int wm   = (wid & 3) * 32;
  const int wn   = (wid >> 2) * 32;
  const int m0   = blockIdx.y * 128;
  const int n0   = blockIdx.x * 64;
  const int l16  = lane >> 4;
  const int lrow = lane & 15;
  const int aloff = l16 * 8;
  const int bkoff = l16 * 8;

  // staging geometry
  int arow[4], acol[4];
#pragma unroll
  for (int i = 0; i < 4; ++i) {
    int idx = tid + i * 256;
    arow[i] = idx >> 3;
    acol[i] = (idx & 7) * 4;
  }
  const int brow = tid >> 3;        // k within slab
  const int bcol = (tid & 7) * 8;   // n within tile

  uint2 areg[4];
  uint4 breg;

  auto g_load = [&](int k0) {
#pragma unroll
    for (int i = 0; i < 4; ++i)
      areg[i] = *(const uint2*)(A + (size_t)(m0 + arow[i]) * lda + k0 + acol[i]);
    breg = *(const uint4*)(B + (size_t)(k0 + brow) * ldb + n0 + bcol);
  };
  auto l_store = [&](int buf) {
#pragma unroll
    for (int i = 0; i < 4; ++i)
      *(uint2*)(&As[buf][arow[i] * 40 + acol[i]]) = areg[i];
    const f16* hp = (const f16*)&breg;
#pragma unroll
    for (int j = 0; j < 8; ++j) Bs[buf][(bcol + j) * 40 + brow] = hp[j];  // transpose
  };

  v8f z = {0.f,0.f,0.f,0.f,0.f,0.f,0.f,0.f};
  v8f c[2][2] = {{z, z}, {z, z}};

  const int nslab = Kdim >> 5;
  g_load(0);
  l_store(0);
  __syncthreads();

  for (int s = 0; s < nslab; ++s) {
    if (s + 1 < nslab) g_load((s + 1) << 5);           // hide behind compute
    if (s + 2 < nslab) {                                // pull tile toward cache
      __builtin_prefetch(A + (size_t)(m0 + arow[0]) * lda + ((s + 2) << 5), 0, 1);
      __builtin_prefetch(B + (size_t)(((s + 2) << 5) + brow) * ldb + n0 + bcol, 0, 1);
    }
    const int buf = s & 1;
    FragAB a[2], bf[2];
#pragma unroll
    for (int mi = 0; mi < 2; ++mi) {
      const f16* ap = &As[buf][(wm + mi * 16 + lrow) * 40];
#pragma unroll
      for (int p = 0; p < 4; ++p) a[mi].u[p] = *(const unsigned*)(ap + 2 * p + aloff);
#pragma unroll
      for (int p = 4; p < 8; ++p) a[mi].u[p] = *(const unsigned*)(ap + 2 * p + 8 + aloff);
    }
#pragma unroll
    for (int ni = 0; ni < 2; ++ni) {
      const uint4* bp = (const uint4*)(&Bs[buf][(wn + ni * 16 + lrow) * 40 + bkoff]);
      ((uint4*)bf[ni].u)[0] = bp[0];
      ((uint4*)bf[ni].u)[1] = bp[2];
    }
#pragma unroll
    for (int mi = 0; mi < 2; ++mi)
#pragma unroll
      for (int ni = 0; ni < 2; ++ni)
        c[mi][ni] = __builtin_amdgcn_wmma_f32_16x16x32_f16(
            false, a[mi].v, false, bf[ni].v, (short)0, c[mi][ni], false, false);
    if (s + 1 < nslab) l_store(buf ^ 1);
    __syncthreads();
  }

  // epilogue
#pragma unroll
  for (int mi = 0; mi < 2; ++mi) {
#pragma unroll
    for (int ni = 0; ni < 2; ++ni) {
#pragma unroll
      for (int j = 0; j < 8; ++j) {
        int row = m0 + wm + mi * 16 + j + 8 * l16;
        int col = n0 + wn + ni * 16 + lrow;
        float acc = c[mi][ni][j];
        if (mode == 0) {
          outh[(size_t)row * ldo + col] = (f16)acc;
        } else if (mode == 1) {
          outf[(size_t)row * ldo + col] =
              acc + bias[col] + resid[(size_t)row * ldo + col];
        } else if (mode == 2) {
          float v = acc + bias[col];
          float t = 0.7978845608028654f * (v + 0.044715f * v * v * v);
          float gl = 0.5f * v * (1.0f + tanhf(t));
          int e = col >> 11;
          outh[(size_t)row * ldo + col] = (f16)(gl * combine[row * 4 + e]);
        } else {
          float v = acc;
#pragma unroll
          for (int e = 0; e < 4; ++e)
            v += combine[row * 4 + e] * b2[e * DMODEL + col];
          outf[(size_t)row * ldo + col] = v + resid[(size_t)row * ldo + col];
        }
      }
    }
  }
}

// ---------------- attention: one wave per (b,h) ----------------
__global__ __launch_bounds__(256)
void attn_kernel(const f16* __restrict__ qkv, f16* __restrict__ attc) {
  __shared__ float Sh[8][32][33];
  __shared__ f16   Ph[8][32][33];
  const int tid  = threadIdx.x;
  const int lane = tid & 31;
  const int wid  = tid >> 5;
  const int pair = blockIdx.x * 8 + wid;
  const int b = pair >> 4;
  const int h = pair & 15;
  const int l16 = lane >> 4;
  const int lrow = lane & 15;
  const int aloff = l16 * 8;
  const size_t base = (size_t)b * 32 * 1536 + h * 32;
  const float scale = 0.044194173824159216f; // 1/sqrt(512)

  v8f z = {0.f,0.f,0.f,0.f,0.f,0.f,0.f,0.f};

  // K^T B-frags: lane -> col s, elements -> e = l16*8 + {0..7, 16..23}
  FragAB kb[2];
#pragma unroll
  for (int st = 0; st < 2; ++st) {
    const uint4* kp = (const uint4*)(qkv + base + (size_t)(st * 16 + lrow) * 1536 + 512 + l16 * 8);
    ((uint4*)kb[st].u)[0] = kp[0];
    ((uint4*)kb[st].u)[1] = kp[2];
  }
  // Q A-frags
  FragAB qa[2];
#pragma unroll
  for (int mt = 0; mt < 2; ++mt) {
    const f16* qp = qkv + base + (size_t)(mt * 16 + lrow) * 1536;
#pragma unroll
    for (int p = 0; p < 4; ++p) qa[mt].u[p] = *(const unsigned*)(qp + 2 * p + aloff);
#pragma unroll
    for (int p = 4; p < 8; ++p) qa[mt].u[p] = *(const unsigned*)(qp + 2 * p + 8 + aloff);
  }
  // 3 live score tiles (tile (0,16) fully causal-masked)
  v8f s00 = __builtin_amdgcn_wmma_f32_16x16x32_f16(false, qa[0].v, false, kb[0].v, (short)0, z, false, false);
  v8f s10 = __builtin_amdgcn_wmma_f32_16x16x32_f16(false, qa[1].v, false, kb[0].v, (short)0, z, false, false);
  v8f s11 = __builtin_amdgcn_wmma_f32_16x16x32_f16(false, qa[1].v, false, kb[1].v, (short)0, z, false, false);
#pragma unroll
  for (int j = 0; j < 8; ++j) {
    int m = j + 8 * l16;
    Sh[wid][m][lrow]           = s00[j] * scale;
    Sh[wid][16 + m][lrow]      = s10[j] * scale;
    Sh[wid][16 + m][16 + lrow] = s11[j] * scale;
  }
  __syncthreads();

  // softmax: lane t handles row t, cols 0..t
  {
    int t = lane;
    float mx = -3.0e38f;
    for (int s = 0; s <= t; ++s) mx = fmaxf(mx, Sh[wid][t][s]);
    float sum = 0.f;
    for (int s = 0; s <= t; ++s) sum += __expf(Sh[wid][t][s] - mx);
    float inv = 1.0f / sum;
    for (int s = 0; s < 32; ++s)
      Ph[wid][t][s] = (s <= t) ? (f16)(__expf(Sh[wid][t][s] - mx) * inv) : (f16)0.0f;
  }
  __syncthreads();

  // V B-frags (strided in s -> scalar gather)
  FragAB vb[2];
#pragma unroll
  for (int et = 0; et < 2; ++et)
#pragma unroll
    for (int i = 0; i < 16; ++i)
      vb[et].h[i] = qkv[base + (size_t)(((i < 8) ? i : i + 8) + l16 * 8) * 1536 + 1024 + et * 16 + lrow];
  // P A-frags from LDS
  FragAB pa[2];
#pragma unroll
  for (int mt = 0; mt < 2; ++mt) {
    int t = mt * 16 + lrow;
#pragma unroll
    for (int i = 0; i < 16; ++i) {
      int s = ((i < 8) ? i : i + 8) + aloff;
      pa[mt].h[i] = Ph[wid][t][s];
    }
  }
#pragma unroll
  for (int mt = 0; mt < 2; ++mt)
#pragma unroll
    for (int et = 0; et < 2; ++et) {
      v8f o = __builtin_amdgcn_wmma_f32_16x16x32_f16(false, pa[mt].v, false, vb[et].v, (short)0, z, false, false);
#pragma unroll
      for (int j = 0; j < 8; ++j) {
        int m = mt * 16 + j + 8 * l16;
        attc[(size_t)(b * 32 + m) * DMODEL + h * 32 + et * 16 + lrow] = (f16)o[j];
      }
    }
}

// ---------------- gating: top-2 of 4, softmax, balance partial ----------------
__global__ __launch_bounds__(256)
void gate_kernel(const f16* __restrict__ h2, const float* __restrict__ wg,
                 float* __restrict__ combine, float* __restrict__ balpart) {
  __shared__ float red[256];
  int tid = threadIdx.x;
  int n = blockIdx.x * 256 + tid;
  const f16* hr = h2 + (size_t)n * DMODEL;
  float a0 = 0.f, a1 = 0.f, a2 = 0.f, a3 = 0.f;
  for (int d = 0; d < DMODEL; ++d) {
    float hv = (float)hr[d];
    const float* wr = wg + d * 4;
    a0 += hv * wr[0]; a1 += hv * wr[1]; a2 += hv * wr[2]; a3 += hv * wr[3];
  }
  float vv[4] = {a0, a1, a2, a3};
  int i0 = 0; float v0 = vv[0];
  for (int e = 1; e < 4; ++e) if (vv[e] > v0) { v0 = vv[e]; i0 = e; }
  int i1 = -1; float v1 = -3.0e38f;
  for (int e = 0; e < 4; ++e) if (e != i0 && vv[e] > v1) { v1 = vv[e]; i1 = e; }
  float e1 = __expf(v1 - v0), s = 1.0f + e1;
  float p0 = 1.0f / s, p1 = e1 / s;
  for (int e = 0; e < 4; ++e)
    combine[(size_t)n * 4 + e] = (e == i0) ? p0 : ((e == i1) ? p1 : 0.0f);
  red[tid] = a0 + a1 + a2 + a3; __syncthreads();
  for (int st = 128; st > 0; st >>= 1) { if (tid < st) red[tid] += red[tid + st]; __syncthreads(); }
  if (tid == 0) balpart[blockIdx.x] = red[0];
}

__global__ void bal_final(const float* __restrict__ balpart, float* __restrict__ out) {
  __shared__ float red[64];
  int tid = threadIdx.x;
  red[tid] = balpart[tid]; __syncthreads();
  for (int s = 32; s > 0; s >>= 1) { if (tid < s) red[tid] += red[tid + s]; __syncthreads(); }
  if (tid == 0) {
    float ep = red[0] / 65536.0f;     // mean over N*E logits
    out[0] = ep * logf(ep + 0.1f);
  }
}

// ---------------- launcher ----------------
extern "C" void kernel_launch(void* const* d_in, const int* in_sizes, int n_in,
                              void* d_out, int out_size, void* d_ws, size_t ws_size,
                              hipStream_t stream) {
  const float* x      = (const float*)d_in[0];
  const float* ln1_g  = (const float*)d_in[1];
  const float* ln1_b  = (const float*)d_in[2];
  const float* wq     = (const float*)d_in[3];
  const float* wk     = (const float*)d_in[4];
  const float* wv     = (const float*)d_in[5];
  const float* w_proj = (const float*)d_in[6];
  const float* b_proj = (const float*)d_in[7];
  const float* ln2_g  = (const float*)d_in[8];
  const float* ln2_b  = (const float*)d_in[9];
  const float* w_gate = (const float*)d_in[10];
  const float* w1     = (const float*)d_in[11];
  const float* b1     = (const float*)d_in[12];
  const float* w2     = (const float*)d_in[13];
  const float* b2     = (const float*)d_in[14];
  float* out = (float*)d_out;

  char* ws = (char*)d_ws;
  size_t off = 0;
  auto alloc = [&](size_t bytes) { void* p = ws + off; off = (off + bytes + 255) & ~(size_t)255; return p; };

  f16*   h16     = (f16*)alloc((size_t)NROWS * DMODEL * 2);
  f16*   wqkv16  = (f16*)alloc((size_t)DMODEL * 1536 * 2);
  f16*   wproj16 = (f16*)alloc((size_t)DMODEL * DMODEL * 2);
  f16*   w1_16   = (f16*)alloc((size_t)DMODEL * KMOE * 2);
  f16*   w2_16   = (f16*)alloc((size_t)KMOE * DMODEL * 2);
  f16*   qkv16   = (f16*)alloc((size_t)NROWS * 1536 * 2);
  f16*   attc16  = (f16*)alloc((size_t)NROWS * DMODEL * 2);
  float* x1      = (float*)alloc((size_t)NROWS * DMODEL * 4);
  f16*   h2_16   = (f16*)alloc((size_t)NROWS * DMODEL * 2);
  float* combine = (float*)alloc((size_t)NROWS * 4 * 4);
  float* balpart = (float*)alloc(64 * 4);
  f16*   mid16   = (f16*)alloc((size_t)CHUNK * KMOE * 2);
  (void)ws_size; (void)in_sizes; (void)n_in; (void)out_size;

  // weight repack
  cvt_qkv<<<(512 * 1536) / 256, 256, 0, stream>>>(wq, wk, wv, wqkv16);
  cvt_f32_f16<<<(512 * 512) / 256, 256, 0, stream>>>(w_proj, wproj16, 512 * 512);
  cvt_w1<<<(512 * 8192) / 256, 256, 0, stream>>>(w1, w1_16);
  cvt_f32_f16<<<(8192 * 512) / 256, 256, 0, stream>>>(w2, w2_16, 8192 * 512);

  // LN1 -> h16
  ln_kernel<<<NROWS, 256, 0, stream>>>(x, ln1_g, ln1_b, h16);

  // QKV: [16384,512] x [512,1536]
  gemm_wmma<<<dim3(1536 / 64, NROWS / 128), 256, 0, stream>>>(
      h16, DMODEL, wqkv16, 1536, DMODEL, 0, qkv16, nullptr, 1536,
      nullptr, nullptr, nullptr, nullptr);

  // attention
  attn_kernel<<<(512 * 16) / 8, 256, 0, stream>>>(qkv16, attc16);

  // out-proj + bias + residual
  gemm_wmma<<<dim3(DMODEL / 64, NROWS / 128), 256, 0, stream>>>(
      attc16, DMODEL, wproj16, DMODEL, DMODEL, 1, nullptr, x1, DMODEL,
      b_proj, x, nullptr, nullptr);

  // LN2 -> h2
  ln_kernel<<<NROWS, 256, 0, stream>>>(x1, ln2_g, ln2_b, h2_16);

  // gating + balance loss
  gate_kernel<<<NROWS / 256, 256, 0, stream>>>(h2_16, w_gate, combine, balpart);
  bal_final<<<1, 64, 0, stream>>>(balpart, out + (size_t)NROWS * DMODEL);

  // MoE, chunked over rows; gate weight folded into mid -> single K=8192 GEMM2
  for (int c = 0; c < NCHUNK; ++c) {
    const size_t r0 = (size_t)c * CHUNK;
    gemm_wmma<<<dim3(KMOE / 64, CHUNK / 128), 256, 0, stream>>>(
        h2_16 + r0 * DMODEL, DMODEL, w1_16, KMOE, DMODEL, 2,
        mid16, nullptr, KMOE, b1, nullptr, combine + r0 * 4, nullptr);
    gemm_wmma<<<dim3(DMODEL / 64, CHUNK / 128), 256, 0, stream>>>(
        mid16, KMOE, w2_16, DMODEL, KMOE, 3,
        nullptr, out + r0 * DMODEL, DMODEL, nullptr,
        x1 + r0 * DMODEL, combine + r0 * 4, b2);
  }
}
